// NOVAPointCloudTransformer_66365834658208
// MI455X (gfx1250) — matrix operand order repaired and checked by reading, not compiled, not run on hardware
//
#include <hip/hip_runtime.h>
#include <hip/hip_bf16.h>

typedef __attribute__((ext_vector_type(16))) _Float16 v16h;
typedef __attribute__((ext_vector_type(2)))  _Float16 v2h;
typedef __attribute__((ext_vector_type(8)))  float    v8f;

#define NBATCH 4
#define NPTS   8192
#define MPTS   1024
#define DIM    768
#define NHEAD  12
#define DH     64

// ---------------------------------------------------------------- WMMA helpers

__device__ __forceinline__ v8f wmma16(v16h a, v16h b, v8f c) {
  // D = A(16x32 f16) * B(32x16 f16) + C(16x16 f32)
  return __builtin_amdgcn_wmma_f32_16x16x32_f16(false, a, false, b, (short)0, c,
                                                false, false);
}

// A fragment: 16x32 f16, A row-major [*, lda]. Contiguous k (not ISA order): lane%16 = row,
// lane/16 selects K-half (+16), VGPR v covers K = 2v (+e).
__device__ __forceinline__ v16h load_a_frag(const _Float16* A, int lda,
                                            int row0, int k0, int lane) {
  const int grp = lane >> 4;
  const _Float16* base = A + (size_t)(row0 + (lane & 15)) * lda + k0 + grp * 16;
  v16h a;
#pragma unroll
  for (int v = 0; v < 8; ++v) {
    v2h p = *(const v2h*)(base + 2 * v);
    a[2 * v]     = p.x;
    a[2 * v + 1] = p.y;
  }
  return a;
}

// B fragment: 32x16 f16 of logical B[k][n], supplied as Bt row-major [N,K]
// (Bt[n*ldb + k] == B[k][n]). Contiguous k, as A: lane%16 = col, lane/16 = K-half (+16),
// VGPR v covers K = 2v (+e).
__device__ __forceinline__ v16h load_bt_frag(const _Float16* Bt, int ldb,
                                             int k0, int n0, int lane) {
  const int grp = lane >> 4;
  const _Float16* base = Bt + (size_t)(n0 + (lane & 15)) * ldb + k0 + grp * 16;
  v16h b;
#pragma unroll
  for (int v = 0; v < 8; ++v) {
    v2h p = *(const v2h*)(base + 2 * v);
    b[2 * v]     = p.x;
    b[2 * v + 1] = p.y;
  }
  return b;
}

// ------------------------------------------------------------ 1) FPS (serial)

__global__ __launch_bounds__(1024) void fps_kernel(const float* __restrict__ pts,
                                                   float* __restrict__ sampled) {
  const int b = blockIdx.x;
  const int tid = threadIdx.x;
  const float* P = pts + (size_t)b * NPTS * 3;

  float px[8], py[8], pz[8], md[8];
#pragma unroll
  for (int j = 0; j < 8; ++j) {
    int i = tid * 8 + j;
    px[j] = P[i * 3 + 0];
    py[j] = P[i * 3 + 1];
    pz[j] = P[i * 3 + 2];
  }
  float cx = P[0], cy = P[1], cz = P[2];
  if (tid == 0) {
    sampled[(size_t)b * MPTS * 3 + 0] = cx;
    sampled[(size_t)b * MPTS * 3 + 1] = cy;
    sampled[(size_t)b * MPTS * 3 + 2] = cz;
  }
#pragma unroll
  for (int j = 0; j < 8; ++j) {
    float dx = px[j] - cx, dy = py[j] - cy, dz = pz[j] - cz;
    md[j] = dx * dx + dy * dy + dz * dz;
  }

  __shared__ float s_val[32];
  __shared__ int   s_idx[32];
  __shared__ float s_cur[3];

  for (int t = 1; t < MPTS; ++t) {
    float bv = -1.0f;
    int   bi = 0;
#pragma unroll
    for (int j = 0; j < 8; ++j) {
      int i = tid * 8 + j;
      if (md[j] > bv) { bv = md[j]; bi = i; }
    }
#pragma unroll
    for (int off = 16; off > 0; off >>= 1) {
      float ov = __shfl_xor(bv, off, 32);
      int   oi = __shfl_xor(bi, off, 32);
      if (ov > bv || (ov == bv && oi < bi)) { bv = ov; bi = oi; }
    }
    if ((tid & 31) == 0) { s_val[tid >> 5] = bv; s_idx[tid >> 5] = bi; }
    __syncthreads();
    if (tid < 32) {
      bv = s_val[tid];
      bi = s_idx[tid];
#pragma unroll
      for (int off = 16; off > 0; off >>= 1) {
        float ov = __shfl_xor(bv, off, 32);
        int   oi = __shfl_xor(bi, off, 32);
        if (ov > bv || (ov == bv && oi < bi)) { bv = ov; bi = oi; }
      }
      if (tid == 0) {
        float wx = P[bi * 3 + 0], wy = P[bi * 3 + 1], wz = P[bi * 3 + 2];
        s_cur[0] = wx; s_cur[1] = wy; s_cur[2] = wz;
        size_t o = ((size_t)b * MPTS + t) * 3;
        sampled[o + 0] = wx; sampled[o + 1] = wy; sampled[o + 2] = wz;
      }
    }
    __syncthreads();
    cx = s_cur[0]; cy = s_cur[1]; cz = s_cur[2];
#pragma unroll
    for (int j = 0; j < 8; ++j) {
      float dx = px[j] - cx, dy = py[j] - cy, dz = pz[j] - cz;
      float d = dx * dx + dy * dy + dz * dz;
      md[j] = fminf(md[j], d);
    }
  }
}

// --------------------------------------------- 2) kNN (k=8) + centroid delta

__global__ __launch_bounds__(1024) void knn_delta_kernel(const float* __restrict__ sampled,
                                                         float* __restrict__ delta) {
  __shared__ float sx[MPTS], sy[MPTS], sz[MPTS];
  const int b = blockIdx.x, q = threadIdx.x;
  const float* S = sampled + (size_t)b * MPTS * 3;
  sx[q] = S[q * 3 + 0];
  sy[q] = S[q * 3 + 1];
  sz[q] = S[q * 3 + 2];
  __syncthreads();
  const float qx = sx[q], qy = sy[q], qz = sz[q];

  float nd[8];
  int   ni[8];
#pragma unroll
  for (int j = 0; j < 8; ++j) { nd[j] = 3.0e38f; ni[j] = 0; }

  for (int n = 0; n < MPTS; ++n) {
    float dx = sx[n] - qx, dy = sy[n] - qy, dz = sz[n] - qz;
    float d = dx * dx + dy * dy + dz * dz;
    if (d < nd[7]) {
      nd[7] = d; ni[7] = n;
#pragma unroll
      for (int j = 7; j > 0; --j) {
        if (nd[j] < nd[j - 1]) {
          float tv = nd[j]; nd[j] = nd[j - 1]; nd[j - 1] = tv;
          int   ti = ni[j]; ni[j] = ni[j - 1]; ni[j - 1] = ti;
        }
      }
    }
  }
  float ax = 0.f, ay = 0.f, az = 0.f;
#pragma unroll
  for (int j = 0; j < 8; ++j) { ax += sx[ni[j]]; ay += sy[ni[j]]; az += sz[ni[j]]; }
  size_t o = ((size_t)b * MPTS + q) * 3;
  delta[o + 0] = qx - ax * 0.125f;
  delta[o + 1] = qy - ay * 0.125f;
  delta[o + 2] = qz - az * 0.125f;
}

// -------------------------- 3) small VALU kernels (K=3 GEMMs, weight convert)

// edge_f16[m,d] = delta[m,:]·W_embed[:,d]   (bias cancels algebraically)
__global__ void edge_kernel(const float* __restrict__ delta,
                            const float* __restrict__ W,
                            _Float16* __restrict__ edge) {
  int gid = blockIdx.x * 256 + threadIdx.x;
  int m = gid / DIM, d = gid % DIM;
  float v = delta[m * 3 + 0] * W[d] + delta[m * 3 + 1] * W[DIM + d] +
            delta[m * 3 + 2] * W[2 * DIM + d];
  edge[gid] = (_Float16)v;
}

// out[m,n] = sampled[m,:]·Ws[:,n] + bs[n] + bo[n]   (final GEMM accumulates on top)
__global__ void spatial_kernel(const float* __restrict__ sampled,
                               const float* __restrict__ Ws,
                               const float* __restrict__ bs,
                               const float* __restrict__ bo,
                               float* __restrict__ out) {
  int gid = blockIdx.x * 256 + threadIdx.x;
  int m = gid / DIM, n = gid % DIM;
  out[gid] = sampled[m * 3 + 0] * Ws[n] + sampled[m * 3 + 1] * Ws[DIM + n] +
             sampled[m * 3 + 2] * Ws[2 * DIM + n] + bs[n] + bo[n];
}

// Transpose + f32->f16 the four DxD weights: Wt[w][n*768+k] = W[w][k*768+n]
__global__ void convert_wt_kernel(const float* __restrict__ Wq,
                                  const float* __restrict__ Wk,
                                  const float* __restrict__ Wv,
                                  const float* __restrict__ Wo,
                                  _Float16* __restrict__ out) {
  int gid = blockIdx.x * 256 + threadIdx.x;
  int w = gid / (DIM * DIM);
  int r = gid % (DIM * DIM);
  int n = r / DIM, k = r % DIM;
  const float* W = (w == 0) ? Wq : (w == 1) ? Wk : (w == 2) ? Wv : Wo;
  out[(size_t)w * DIM * DIM + (size_t)n * DIM + k] = (_Float16)W[(size_t)k * DIM + n];
}

// ------------------------------------- 4) WMMA projection GEMM (Q / K / Vt)

// C[4096, 768] = A[4096,768] x B[768,768] + bias, stored f16 either as
// [B,H,S,dh] (mode 0: Q,K) or [B,H,dh,S] (mode 1: V transposed).
__global__ __launch_bounds__(256) void gemm_qkv_kernel(const _Float16* __restrict__ A,
                                                       const _Float16* __restrict__ Bt,
                                                       const float* __restrict__ bias,
                                                       _Float16* __restrict__ out,
                                                       int mode) {
  const int lane = threadIdx.x & 31;
  const int wave = threadIdx.x >> 5;
  const int m0 = blockIdx.y * 128 + wave * 16;
  const int n0 = blockIdx.x * 64;

  v8f acc[4] = {};
  for (int k0 = 0; k0 < DIM; k0 += 32) {
    v16h a = load_a_frag(A, DIM, m0, k0, lane);
#pragma unroll
    for (int t = 0; t < 4; ++t) {
      v16h b = load_bt_frag(Bt, DIM, k0, n0 + t * 16, lane);
      acc[t] = wmma16(a, b, acc[t]);
    }
  }
  const int grp = lane >> 4, col = lane & 15;
#pragma unroll
  for (int t = 0; t < 4; ++t) {
    int n = n0 + t * 16 + col;
    float bi = bias[n];
    int h = n >> 6, d = n & 63;
#pragma unroll
    for (int v = 0; v < 8; ++v) {
      int m = m0 + v + grp * 8;
      int bb = m >> 10, s = m & 1023;
      float val = acc[t][v] + bi;
      if (mode == 0)
        out[(((size_t)(bb * NHEAD + h)) * MPTS + s) * DH + d] = (_Float16)val;
      else
        out[(((size_t)(bb * NHEAD + h)) * DH + d) * MPTS + s] = (_Float16)val;
    }
  }
}

// --------------------------------- 5) flash attention (wave = 16 query rows)

__global__ __launch_bounds__(256) void attention_kernel(const _Float16* __restrict__ Q,
                                                        const _Float16* __restrict__ K,
                                                        const _Float16* __restrict__ Vt,
                                                        _Float16* __restrict__ O) {
  __shared__ _Float16 sP[8][16 * 32];  // per-wave P staging (C-layout -> A-layout)
  const int lane = threadIdx.x & 31;
  const int wave = threadIdx.x >> 5;
  const int w = blockIdx.x * 8 + wave;
  const int b = w / (NHEAD * 64);
  const int rem = w % (NHEAD * 64);
  const int h = rem / 64;
  const int qb = rem % 64;

  const _Float16* Qbh = Q + ((size_t)(b * NHEAD + h)) * MPTS * DH;
  const _Float16* Kbh = K + ((size_t)(b * NHEAD + h)) * MPTS * DH;
  const _Float16* Vbh = Vt + ((size_t)(b * NHEAD + h)) * DH * MPTS;
  const int q0 = qb * 16;
  const int grp = lane >> 4, col = lane & 15;

  v16h aq0 = load_a_frag(Qbh, DH, q0, 0, lane);
  v16h aq1 = load_a_frag(Qbh, DH, q0, 32, lane);

  v8f acc[4] = {};
  float mrow[8], lrow[8];
#pragma unroll
  for (int v = 0; v < 8; ++v) { mrow[v] = -1.0e30f; lrow[v] = 0.f; }
  _Float16* myP = sP[wave];

  for (int kb = 0; kb < MPTS; kb += 32) {
    // S = (Q Kt) * scale for 32 keys -> two 16x16 f32 tiles
    v8f s0 = {}, s1 = {};
    {
      v16h bk = load_bt_frag(Kbh, DH, 0, kb, lane);
      s0 = wmma16(aq0, bk, s0);
      bk = load_bt_frag(Kbh, DH, 32, kb, lane);
      s0 = wmma16(aq1, bk, s0);
      bk = load_bt_frag(Kbh, DH, 0, kb + 16, lane);
      s1 = wmma16(aq0, bk, s1);
      bk = load_bt_frag(Kbh, DH, 32, kb + 16, lane);
      s1 = wmma16(aq1, bk, s1);
    }
    float mnew[8], rs[8];
#pragma unroll
    for (int v = 0; v < 8; ++v) {
      float a0 = s0[v] * 0.125f, a1 = s1[v] * 0.125f;
      s0[v] = a0; s1[v] = a1;
      float mx = fmaxf(a0, a1);
      mx = fmaxf(mx, __shfl_xor(mx, 1, 32));
      mx = fmaxf(mx, __shfl_xor(mx, 2, 32));
      mx = fmaxf(mx, __shfl_xor(mx, 4, 32));
      mx = fmaxf(mx, __shfl_xor(mx, 8, 32));
      mnew[v] = fmaxf(mrow[v], mx);
    }
#pragma unroll
    for (int v = 0; v < 8; ++v) {
      float p0 = __expf(s0[v] - mnew[v]);
      float p1 = __expf(s1[v] - mnew[v]);
      myP[(v + 8 * grp) * 32 + col]      = (_Float16)p0;
      myP[(v + 8 * grp) * 32 + col + 16] = (_Float16)p1;
      float r = p0 + p1;
      r += __shfl_xor(r, 1, 32);
      r += __shfl_xor(r, 2, 32);
      r += __shfl_xor(r, 4, 32);
      r += __shfl_xor(r, 8, 32);
      rs[v] = r;
    }
#pragma unroll
    for (int v = 0; v < 8; ++v) {
      float alpha = __expf(mrow[v] - mnew[v]);
      lrow[v] = lrow[v] * alpha + rs[v];
      mrow[v] = mnew[v];
#pragma unroll
      for (int t = 0; t < 4; ++t) acc[t][v] *= alpha;
    }
    // same-wave LDS RAW: DS ops are in-order, wait for data return
    asm volatile("s_wait_dscnt 0" ::: "memory");
    v16h ap = load_a_frag(myP, 32, 0, 0, lane);
#pragma unroll
    for (int t = 0; t < 4; ++t) {
      v16h bv = load_bt_frag(Vbh, MPTS, kb, t * 16, lane);
      acc[t] = wmma16(ap, bv, acc[t]);
    }
  }
  // O[b*1024+s][h*64+d] = acc / l
#pragma unroll
  for (int t = 0; t < 4; ++t) {
#pragma unroll
    for (int v = 0; v < 8; ++v) {
      int m = b * MPTS + q0 + v + 8 * grp;
      int n = h * DH + t * 16 + col;
      O[(size_t)m * DIM + n] = (_Float16)(acc[t][v] / lrow[v]);
    }
  }
}

// ----------------------------- 6) output projection: d_out += attnO x Wo

__global__ __launch_bounds__(256) void gemm_out_kernel(const _Float16* __restrict__ A,
                                                       const _Float16* __restrict__ Bt,
                                                       float* __restrict__ C) {
  const int lane = threadIdx.x & 31;
  const int wave = threadIdx.x >> 5;
  const int m0 = blockIdx.y * 128 + wave * 16;
  const int n0 = blockIdx.x * 64;

  v8f acc[4] = {};
  for (int k0 = 0; k0 < DIM; k0 += 32) {
    v16h a = load_a_frag(A, DIM, m0, k0, lane);
#pragma unroll
    for (int t = 0; t < 4; ++t) {
      v16h b = load_bt_frag(Bt, DIM, k0, n0 + t * 16, lane);
      acc[t] = wmma16(a, b, acc[t]);
    }
  }
  const int grp = lane >> 4, col = lane & 15;
#pragma unroll
  for (int t = 0; t < 4; ++t) {
#pragma unroll
    for (int v = 0; v < 8; ++v) {
      size_t o = (size_t)(m0 + v + grp * 8) * DIM + n0 + t * 16 + col;
      C[o] += acc[t][v];
    }
  }
}

// ------------------------------------------------------------------- launch

extern "C" void kernel_launch(void* const* d_in, const int* in_sizes, int n_in,
                              void* d_out, int out_size, void* d_ws, size_t ws_size,
                              hipStream_t stream) {
  (void)in_sizes; (void)n_in; (void)out_size; (void)ws_size;
  const float* points  = (const float*)d_in[0];
  const float* W_embed = (const float*)d_in[1];
  // b_embed (d_in[2]) cancels in edge features
  const float* Wq = (const float*)d_in[3];
  const float* bq = (const float*)d_in[4];
  const float* Wk = (const float*)d_in[5];
  const float* bk = (const float*)d_in[6];
  const float* Wv = (const float*)d_in[7];
  const float* bv = (const float*)d_in[8];
  const float* Wo = (const float*)d_in[9];
  const float* bo = (const float*)d_in[10];
  const float* Ws = (const float*)d_in[11];
  const float* bs = (const float*)d_in[12];
  float* out = (float*)d_out;

  char* ws = (char*)d_ws;
  const size_t MB2 = (size_t)NBATCH * MPTS * DIM * 2;     // 6291456 (f16 [4096,768])
  float*    sampled = (float*)(ws);                        // 49152 B
  float*    delta   = (float*)(ws + 49152);                // 49152 B
  _Float16* edge    = (_Float16*)(ws + 98304);             // MB2
  _Float16* Wt      = (_Float16*)(ws + 98304 + MB2);       // 4x 1179648 B
  _Float16* Qb      = (_Float16*)(ws + 98304 + MB2 + 4718592);
  _Float16* Kb      = (_Float16*)((char*)Qb + MB2);
  _Float16* Vtb     = (_Float16*)((char*)Kb + MB2);
  _Float16* attnO   = (_Float16*)((char*)Vtb + MB2);
  _Float16* Wqt = Wt;
  _Float16* Wkt = Wt + (size_t)DIM * DIM;
  _Float16* Wvt = Wt + (size_t)2 * DIM * DIM;
  _Float16* Wot = Wt + (size_t)3 * DIM * DIM;

  // 1) FPS
  fps_kernel<<<NBATCH, 1024, 0, stream>>>(points, sampled);
  // 2) kNN + centroid delta
  knn_delta_kernel<<<NBATCH, 1024, 0, stream>>>(sampled, delta);
  // 3) weight convert/transpose + edge + spatial(+biases) into d_out
  convert_wt_kernel<<<(4 * DIM * DIM) / 256, 256, 0, stream>>>(Wq, Wk, Wv, Wo, Wt);
  edge_kernel<<<(NBATCH * MPTS * DIM) / 256, 256, 0, stream>>>(delta, W_embed, edge);
  spatial_kernel<<<(NBATCH * MPTS * DIM) / 256, 256, 0, stream>>>(sampled, Ws, bs, bo, out);
  // 4) Q/K/V projections (WMMA)
  dim3 gproj(DIM / 64, (NBATCH * MPTS) / 128);
  gemm_qkv_kernel<<<gproj, 256, 0, stream>>>(edge, Wqt, bq, Qb, 0);
  gemm_qkv_kernel<<<gproj, 256, 0, stream>>>(edge, Wkt, bk, Kb, 0);
  gemm_qkv_kernel<<<gproj, 256, 0, stream>>>(edge, Wvt, bv, Vtb, 1);
  // 5) attention: 4*12*64 waves, 8 waves/block
  attention_kernel<<<(NBATCH * NHEAD * 64) / 8, 256, 0, stream>>>(Qb, Kb, Vtb, attnO);
  // 6) output projection accumulated onto spatial+biases
  gemm_out_kernel<<<gproj, 256, 0, stream>>>(attnO, Wot, out);
}
